// MutiHeadAttention_47098611368037
// MI455X (gfx1250) — matrix operand order repaired and checked by reading, not compiled, not run on hardware
//
#include <hip/hip_runtime.h>

#define NH    16
#define HD    128
#define BATCH 4
#define SEQ   2048
#define HID   2048

typedef __attribute__((ext_vector_type(16))) __bf16 v16bf;
typedef __attribute__((ext_vector_type(8)))  float  v8f;
typedef __attribute__((ext_vector_type(4)))  int    v4i_t;
typedef __attribute__((address_space(1))) v4i_t g_v4i;   // global int4*
typedef __attribute__((address_space(3))) v4i_t l_v4i;   // LDS int4*

#if defined(__has_builtin)
#if __has_builtin(__builtin_amdgcn_global_load_async_to_lds_b128)
#define HAVE_ASYNC_LDS 1
#endif
#if __has_builtin(__builtin_amdgcn_cvt_pk_bf16_f32)
#define HAVE_PK_BF16 1
#endif
#endif

union Frag16 {                     // one 16x32 (A) or 32x16 (B) bf16 WMMA operand
  v16bf          v;
  unsigned short h[16];
  unsigned int   u[8];
  uint4          q[2];
};

__device__ __forceinline__ unsigned short f2bf(float f) {
  unsigned int u = __float_as_uint(f);
  u += 0x7FFFu + ((u >> 16) & 1u);   // round-to-nearest-even
  return (unsigned short)(u >> 16);
}

// pack two f32 -> two bf16 in one dword (v_cvt_pk_bf16_f32 when available)
__device__ __forceinline__ unsigned int f2bf_pk(float lo, float hi_) {
#if HAVE_PK_BF16
  auto pk = __builtin_amdgcn_cvt_pk_bf16_f32(lo, hi_);
  union { decltype(pk) p; unsigned int u; } cv;
  cv.p = pk;
  return cv.u;
#else
  return (unsigned int)f2bf(lo) | ((unsigned int)f2bf(hi_) << 16);
#endif
}

__device__ __forceinline__ v8f vzero8() {
  v8f r;
#pragma unroll
  for (int i = 0; i < 8; ++i) r[i] = 0.0f;
  return r;
}

__device__ __forceinline__ v8f wmma_bf16(const Frag16& a, const Frag16& b, v8f c) {
  return __builtin_amdgcn_wmma_f32_16x16x32_bf16(false, a.v, false, b.v,
                                                 (short)0, c, false, false);
}

// DPP16 row_ror rotations (rows of 16 lanes on wave32): pure-VALU reductions.
template <int CTRL>
__device__ __forceinline__ float dpp_rot(float x) {
  return __int_as_float(
      __builtin_amdgcn_update_dpp(0, __float_as_int(x), CTRL, 0xF, 0xF, true));
}
__device__ __forceinline__ float rowmax16(float x) {
  x = fmaxf(x, dpp_rot<0x121>(x));   // row_ror:1
  x = fmaxf(x, dpp_rot<0x122>(x));   // row_ror:2
  x = fmaxf(x, dpp_rot<0x124>(x));   // row_ror:4
  x = fmaxf(x, dpp_rot<0x128>(x));   // row_ror:8
  return x;
}
__device__ __forceinline__ float rowsum16(float x) {
  x += dpp_rot<0x121>(x);
  x += dpp_rot<0x122>(x);
  x += dpp_rot<0x124>(x);
  x += dpp_rot<0x128>(x);
  return x;
}

// ---------------------------------------------------------------------------
// C[M,N] = A[M,K] @ W[N,K]^T + bias[N], LDS double-buffered.
// A: f32 (AT=float) or bf16 raw (AT=unsigned short). Output: bf16 or f32.
// ---------------------------------------------------------------------------
template <typename AT, bool OUT_BF16>
__global__ __launch_bounds__(256) void gemm_bias_kernel(
    const AT* __restrict__ A, const float* __restrict__ W,
    const float* __restrict__ bias, void* __restrict__ out,
    int M, int N, int K) {
  constexpr int LDT = 40;  // padded LDS stride (shorts) -> bank-conflict-free
  __shared__ unsigned short sA[2][128 * LDT];
  __shared__ unsigned short sB[2][128 * LDT];

  const int tid  = threadIdx.x;
  const int lane = tid & 31;
  const int wave = tid >> 5;
  const int hi   = (lane >> 4) & 1;
  const int ln   = lane & 15;
  const int wm   = wave >> 1;        // 0..3 : 32-row stripe
  const int wn   = wave & 1;         // 0..1 : 64-col stripe
  const int m0   = blockIdx.y * 128;
  const int n0   = blockIdx.x * 128;

  const int srow  = tid >> 1;        // staging row 0..127
  const int shalf = (tid & 1) * 16;  // staging k-offset

  auto stage = [&](int buf, int kb) {
    {  // A tile 128x32 -> bf16
      union { unsigned short h[16]; unsigned int u[8]; uint4 q[2]; } stg;
      if constexpr (sizeof(AT) == 4) {
        const float* src = (const float*)A + (size_t)(m0 + srow) * K + kb + shalf;
#pragma unroll
        for (int j = 0; j < 8; ++j) stg.u[j] = f2bf_pk(src[2 * j], src[2 * j + 1]);
      } else {
        const uint4* src = (const uint4*)((const unsigned short*)A +
                                          (size_t)(m0 + srow) * K + kb + shalf);
        stg.q[0] = src[0];
        stg.q[1] = src[1];
      }
      *(uint4*)&sA[buf][srow * LDT + shalf]     = stg.q[0];
      *(uint4*)&sA[buf][srow * LDT + shalf + 8] = stg.q[1];
    }
    {  // W tile 128x32 (N-major) -> bf16
      union { unsigned short h[16]; unsigned int u[8]; uint4 q[2]; } stg;
      const float* src = W + (size_t)(n0 + srow) * K + kb + shalf;
#pragma unroll
      for (int j = 0; j < 8; ++j) stg.u[j] = f2bf_pk(src[2 * j], src[2 * j + 1]);
      *(uint4*)&sB[buf][srow * LDT + shalf]     = stg.q[0];
      *(uint4*)&sB[buf][srow * LDT + shalf + 8] = stg.q[1];
    }
  };

  v8f acc[2][4];
#pragma unroll
  for (int i = 0; i < 2; ++i)
#pragma unroll
    for (int j = 0; j < 4; ++j) acc[i][j] = vzero8();

  stage(0, 0);
  int p = 0;
  for (int kb = 0; kb < K; kb += 32) {
    __syncthreads();
    if (kb + 64 < K) {  // L2 prefetch two tiles ahead
      __builtin_prefetch((const char*)A +
                         ((size_t)(m0 + srow) * K + kb + 64 + shalf) * sizeof(AT), 0, 3);
      __builtin_prefetch(W + (size_t)(n0 + srow) * K + kb + 64 + shalf, 0, 3);
    }
    if (kb + 32 < K) stage(p ^ 1, kb + 32);  // fill other buffer while computing

    // ---- fragments: two contiguous b128 loads per lane (contiguous k = 16*hi + s, same for A and B) ----
    Frag16 a[2];
#pragma unroll
    for (int mt = 0; mt < 2; ++mt) {
      const unsigned short* pr = &sA[p][(wm * 32 + mt * 16 + ln) * LDT];
      a[mt].q[0] = *(const uint4*)&pr[hi * 16];
      a[mt].q[1] = *(const uint4*)&pr[8 + hi * 16];
    }
#pragma unroll
    for (int nt = 0; nt < 4; ++nt) {
      Frag16 b;
      const unsigned short* pr = &sB[p][(wn * 64 + nt * 16 + ln) * LDT];
      b.q[0] = *(const uint4*)&pr[hi * 16];
      b.q[1] = *(const uint4*)&pr[8 + hi * 16];
      acc[0][nt] = wmma_bf16(a[0], b, acc[0][nt]);
      acc[1][nt] = wmma_bf16(a[1], b, acc[1][nt]);
    }
    p ^= 1;
  }

  // ---- epilogue: bias + store ----
#pragma unroll
  for (int mt = 0; mt < 2; ++mt)
#pragma unroll
    for (int nt = 0; nt < 4; ++nt)
#pragma unroll
      for (int e = 0; e < 8; ++e) {
        int row  = m0 + wm * 32 + mt * 16 + e + hi * 8;
        int col  = n0 + wn * 64 + nt * 16 + ln;
        float vv = acc[mt][nt][e] + bias[col];
        if constexpr (OUT_BF16)
          ((unsigned short*)out)[(size_t)row * N + col] = f2bf(vv);
        else
          ((float*)out)[(size_t)row * N + col] = vv;
      }
}

// ---------------------------------------------------------------------------
// Causal flash attention. One wave = one 16-row query tile of one (b,h).
// Q/K frags load straight from global (b128 pairs). V staged via async
// global->LDS (ASYNCcnt) when available, then transposed by a pair-gather.
// ---------------------------------------------------------------------------
__global__ __launch_bounds__(256) void flash_attn_kernel(
    const unsigned short* __restrict__ Qb, const unsigned short* __restrict__ Kb,
    const unsigned short* __restrict__ Vb, unsigned short* __restrict__ Ctx) {
  extern __shared__ unsigned short smem[];
  const int tid  = threadIdx.x;
  const int lane = tid & 31;
  const int wave = tid >> 5;
  const int hi   = (lane >> 4) & 1;
  const int ln   = lane & 15;

  unsigned short* stV = smem + wave * (32 * 136 + 16 * 40);  // V stage, padded
  unsigned short* Pb  = stV + 32 * 136;                      // P transpose buf

  const int tiles = SEQ / 16;  // q tiles per (b,h)
  int g  = blockIdx.x * 8 + wave;
  int qt = g % tiles;
  int h  = (g / tiles) % NH;
  int b  = g / (tiles * NH);
  int q0 = qt * 16;

  const size_t baseBH = (size_t)b * SEQ * HID + (size_t)h * HD;

  // Q A-fragments: 4 chunks of 32 along head dim, direct b128 loads.
  Frag16 qa[4];
  {
    const unsigned short* qrow = Qb + baseBH + (size_t)(q0 + ln) * HID;
#pragma unroll
    for (int c = 0; c < 4; ++c) {
      qa[c].q[0] = *(const uint4*)&qrow[c * 32 + hi * 16];
      qa[c].q[1] = *(const uint4*)&qrow[c * 32 + 8 + hi * 16];
    }
  }

  v8f acc[8];
#pragma unroll
  for (int t = 0; t < 8; ++t) acc[t] = vzero8();
  float mst[8], lst[8];
#pragma unroll
  for (int e = 0; e < 8; ++e) { mst[e] = -1e30f; lst[e] = 0.0f; }

  const float scale = 0.08838834764831845f;  // 1/sqrt(128)
  const int nkb = (q0 + 16 + 31) / 32;       // causal block count (<= SEQ/32)

  for (int kb = 0; kb < nkb; ++kb) {
    const int k0 = kb * 32;

    // ---- stage V block 32x128 -> LDS (lane = kc row) ----
#if HAVE_ASYNC_LDS
    {
      // INST_OFFSET applies to BOTH LDS and global address: fixed base
      // pointers + literal immediate offsets -> zero per-chunk address math.
      g_v4i* gsrc = (g_v4i*)(const_cast<unsigned short*>(Vb) + baseBH +
                             (size_t)(k0 + lane) * HID);
      l_v4i* ldst = (l_v4i*)&stV[lane * 136];
#define ASYNC_CHUNK(OFF) \
  __builtin_amdgcn_global_load_async_to_lds_b128(gsrc, ldst, (OFF), 0)
      ASYNC_CHUNK(0);   ASYNC_CHUNK(16);  ASYNC_CHUNK(32);  ASYNC_CHUNK(48);
      ASYNC_CHUNK(64);  ASYNC_CHUNK(80);  ASYNC_CHUNK(96);  ASYNC_CHUNK(112);
      ASYNC_CHUNK(128); ASYNC_CHUNK(144); ASYNC_CHUNK(160); ASYNC_CHUNK(176);
      ASYNC_CHUNK(192); ASYNC_CHUNK(208); ASYNC_CHUNK(224); ASYNC_CHUNK(240);
#undef ASYNC_CHUNK
    }
#else
    {
      const uint4* src = (const uint4*)(Vb + baseBH + (size_t)(k0 + lane) * HID);
#pragma unroll
      for (int i0 = 0; i0 < 16; i0 += 4) {
        uint4 t0 = src[i0], t1 = src[i0 + 1], t2 = src[i0 + 2], t3 = src[i0 + 3];
        *(uint4*)&stV[lane * 136 + (i0 + 0) * 8] = t0;
        *(uint4*)&stV[lane * 136 + (i0 + 1) * 8] = t1;
        *(uint4*)&stV[lane * 136 + (i0 + 2) * 8] = t2;
        *(uint4*)&stV[lane * 136 + (i0 + 3) * 8] = t3;
      }
    }
#endif

    // ---- S = Q K^T : preload all 8 K B-frags (one clause), then 8 WMMAs ----
    Frag16 bk[8];
#pragma unroll
    for (int nt = 0; nt < 2; ++nt) {
      const unsigned short* kr =
          Kb + baseBH + (size_t)(k0 + nt * 16 + ln) * HID + hi * 16;
#pragma unroll
      for (int c = 0; c < 4; ++c) {
        bk[nt * 4 + c].q[0] = *(const uint4*)&kr[c * 32];
        bk[nt * 4 + c].q[1] = *(const uint4*)&kr[c * 32 + 8];
      }
    }
    v8f s0 = vzero8(), s1 = vzero8();
#pragma unroll
    for (int c = 0; c < 4; ++c) {
      s0 = wmma_bf16(qa[c], bk[c], s0);
      s1 = wmma_bf16(qa[c], bk[4 + c], s1);
    }

    // ---- scale + causal mask + online softmax (DPP16 row reductions) ----
#pragma unroll
    for (int e = 0; e < 8; ++e) {
      int   qg = q0 + e + hi * 8;
      float v0 = s0[e] * scale, v1 = s1[e] * scale;
      if (k0 + ln > qg)      v0 = -1e30f;
      if (k0 + 16 + ln > qg) v1 = -1e30f;
      float rm   = rowmax16(fmaxf(v0, v1));
      float mnew = fmaxf(mst[e], rm);
      float corr = __expf(mst[e] - mnew);
      float p0   = __expf(v0 - mnew);
      float p1   = __expf(v1 - mnew);
      float rs   = rowsum16(p0 + p1);
      lst[e] = lst[e] * corr + rs;
      mst[e] = mnew;
#pragma unroll
      for (int t = 0; t < 8; ++t) acc[t][e] *= corr;
      Pb[(e + hi * 8) * 40 + ln]      = f2bf(p0);
      Pb[(e + hi * 8) * 40 + 16 + ln] = f2bf(p1);
    }

#if HAVE_ASYNC_LDS
    asm volatile("s_wait_asynccnt 0" ::: "memory");  // V block resident in LDS
#endif
    asm volatile("s_wait_dscnt 0" ::: "memory");     // P buffer visible

    // P A-fragment from LDS (C->A layout change done through memory)
    Frag16 pa;
    pa.q[0] = *(const uint4*)&Pb[ln * 40 + hi * 16];
    pa.q[1] = *(const uint4*)&Pb[ln * 40 + 8 + hi * 16];

    // ---- O += P V : gather V^T pairs, 4 frags ahead of each WMMA group ----
#pragma unroll
    for (int tg = 0; tg < 2; ++tg) {
      Frag16 bv[4];
#pragma unroll
      for (int j = 0; j < 4; ++j) {
        const int d = (tg * 4 + j) * 16 + ln;
#pragma unroll
        for (int r = 0; r < 8; ++r) {
          int kc           = 2 * r + hi * 16;
          bv[j].h[2 * r]     = stV[kc * 136 + d];
          bv[j].h[2 * r + 1] = stV[(kc + 1) * 136 + d];
        }
      }
#pragma unroll
      for (int j = 0; j < 4; ++j)
        acc[tg * 4 + j] = wmma_bf16(pa, bv[j], acc[tg * 4 + j]);
    }
  }

  // epilogue: normalize and store context (bf16, [b, q, h*HD + d])
#pragma unroll
  for (int e = 0; e < 8; ++e) {
    float  inv    = 1.0f / lst[e];
    size_t rowoff = baseBH + (size_t)(q0 + e + hi * 8) * HID;
#pragma unroll
    for (int t = 0; t < 8; ++t)
      Ctx[rowoff + t * 16 + ln] = f2bf(acc[t][e] * inv);
  }
}

// ---------------------------------------------------------------------------
extern "C" void kernel_launch(void* const* d_in, const int* in_sizes, int n_in,
                              void* d_out, int out_size, void* d_ws,
                              size_t ws_size, hipStream_t stream) {
  (void)in_sizes; (void)n_in; (void)out_size; (void)ws_size;
  const float* hidden = (const float*)d_in[0];
  // d_in[1] = mask (tril, start_pos==0 -> causal handled analytically)
  const float* Wq = (const float*)d_in[2];
  const float* bq = (const float*)d_in[3];
  const float* Wk = (const float*)d_in[4];
  const float* bk = (const float*)d_in[5];
  const float* Wv = (const float*)d_in[6];
  const float* bv = (const float*)d_in[7];
  const float* Wo = (const float*)d_in[8];
  const float* bo = (const float*)d_in[9];

  const int M = BATCH * SEQ, N = HID, K = HID;
  const size_t elems = (size_t)M * HID;
  unsigned short* qb  = (unsigned short*)d_ws;
  unsigned short* kb  = qb + elems;
  unsigned short* vb  = kb + elems;
  unsigned short* ctx = vb + elems;   // 4 x 32MB bf16 in workspace

  dim3 ggrid(N / 128, M / 128), gblk(256);
  gemm_bias_kernel<float, true><<<ggrid, gblk, 0, stream>>>(hidden, Wq, bq, qb, M, N, K);
  gemm_bias_kernel<float, true><<<ggrid, gblk, 0, stream>>>(hidden, Wk, bk, kb, M, N, K);
  gemm_bias_kernel<float, true><<<ggrid, gblk, 0, stream>>>(hidden, Wv, bv, vb, M, N, K);

  const int smem = 8 * (32 * 136 + 16 * 40) * (int)sizeof(unsigned short); // 79,872 B
  (void)hipFuncSetAttribute((const void*)flash_attn_kernel,
                            hipFuncAttributeMaxDynamicSharedMemorySize, smem);
  const int nwg = (BATCH * NH * (SEQ / 16)) / 8;  // 1024 workgroups, 8 waves each
  flash_attn_kernel<<<dim3(nwg), gblk, smem, stream>>>(qb, kb, vb, ctx);

  gemm_bias_kernel<unsigned short, false><<<ggrid, gblk, 0, stream>>>(
      ctx, Wo, bo, (float*)d_out, M, N, K);
}
